// LiquidModel_52922587021844
// MI455X (gfx1250) — matrix operand order repaired and checked by reading, not compiled, not run on hardware
//
#include <hip/hip_runtime.h>

// Liquid RNN fused kernel for MI455X (gfx1250, wave32, WMMA).
// Single pass over x (134 MB ~ 5.7us HBM floor). Latency-optimized recurrence:
// double-buffered h (1 barrier/step), x-projection software-pipelined one step
// ahead, split WMMA accumulation chains (depth 2 on the critical path).

#define BT    16      // batch tile per workgroup (WMMA M)
#define HD    128     // hidden dim
#define IDIM  64      // input dim
#define ODIM  64      // output dim
#define TSEQ  2048    // sequence length
#define NB    256     // total batch
#define TC    16      // timesteps staged per LDS chunk

#define HPAD  136     // 128 + 8 halves: row stride 272B = 17*16B -> conflict-free b128
#define XPAD  72      // 64 + 8 halves:  row stride 144B = 9*16B  -> conflict-free b128

typedef _Float16 v16h __attribute__((ext_vector_type(16)));
typedef _Float16 v8h  __attribute__((ext_vector_type(8)));
typedef float    v8f  __attribute__((ext_vector_type(8)));

__device__ __forceinline__ float fast_sigmoid(float x) {
  // 1/(1+e^-x); saturation-safe (exp->inf => rcp->0)
  return __builtin_amdgcn_rcpf(1.0f + __expf(-x));
}
__device__ __forceinline__ float fast_tanh(float x) {
  // 1 - 2/(e^{2x}+1); saturates cleanly to +/-1
  return 1.0f - 2.0f * __builtin_amdgcn_rcpf(__expf(2.0f * x) + 1.0f);
}
__device__ __forceinline__ v8f wmma_f16(v16h a, v16h b, v8f c) {
  return __builtin_amdgcn_wmma_f32_16x16x32_f16(false, a, false, b, (short)0, c,
                                                false, false);
}
// A-fragment (16x32 f16) gather: lane holds K {kb..kb+15} of row M
// -> two 16B LDS loads.
__device__ __forceinline__ v16h afrag(const _Float16* p) {
  const v8h lo = *(const v8h*)p;
  const v8h hi = *(const v8h*)(p + 8);
  v16h a;
  #pragma unroll
  for (int i = 0; i < 8; ++i) { a[i] = lo[i]; a[8 + i] = hi[i]; }
  return a;
}

__global__ __launch_bounds__(256, 1)
void liquid_rnn_kernel(const float* __restrict__ x,
                       const float* __restrict__ W_in, const float* __restrict__ b_in,
                       const float* __restrict__ W_st, const float* __restrict__ b_st,
                       const float* __restrict__ W_g,  const float* __restrict__ b_g,
                       const float* __restrict__ W_h,  const float* __restrict__ b_h,
                       float* __restrict__ out)
{
  __shared__ __align__(16) _Float16 h_lds[2][BT][HPAD];   // double-buffered h mirror
  __shared__ __align__(16) _Float16 x_lds[TC][BT][XPAD];

  const int tid  = threadIdx.x;
  const int wave = tid >> 5;        // 0..7 : owns H columns [wave*16, wave*16+16)
  const int lane = tid & 31;
  const int lhal = lane & 15;       // A-row M / C-column N within tile
  const int hiH  = lane >> 4;       // half-wave selector
  const int wg   = blockIdx.x;      // batch tile index (0..15)

  const int n = wave * 16 + lhal;   // H column owned by this lane

  // ---------------- Preload weight B-fragments (fp32 -> f16) ----------------
  // B 32x16 f16 layout: lane = column n; half-waves hold 16 contiguous K each.
  v16h BWgh[4], BWst[4], BWgx[2], BWin[2];
  #pragma unroll
  for (int f = 0; f < 4; ++f) {
    const int kb = f * 32 + hiH * 16;
    const float* pg = W_g  + (size_t)n * (IDIM + HD) + IDIM + kb;  // Wg_h slice
    const float* ps = W_st + (size_t)n * HD + kb;
    #pragma unroll
    for (int i = 0; i < 16; ++i) {
      BWgh[f][i] = (_Float16)pg[i];
      BWst[f][i] = (_Float16)ps[i];
    }
  }
  #pragma unroll
  for (int f = 0; f < 2; ++f) {
    const int kb = f * 32 + hiH * 16;
    const float* pgx = W_g  + (size_t)n * (IDIM + HD) + kb;        // Wg_x slice
    const float* pin = W_in + (size_t)n * IDIM + kb;
    #pragma unroll
    for (int i = 0; i < 16; ++i) {
      BWgx[f][i] = (_Float16)pgx[i];
      BWin[f][i] = (_Float16)pin[i];
    }
  }
  const float bg_n = b_g[n];
  const float bp_n = b_in[n] + b_st[n];

  // ---------------- h = 0 ----------------
  v8f hC = {};                                   // fp32 master copy (C layout)
  for (int i = tid; i < 2 * BT * HPAD; i += 256)
    ((_Float16*)h_lds)[i] = (_Float16)0.0f;
  __syncthreads();

  const int rowM = lhal;                         // A-fragment row
  const int kbA  = hiH * 16;                     // A-fragment lane K base
  int cur = 0;                                   // buffer holding current h

  v8f xg, xp;                                    // look-ahead x-projection (+bias)

  // ---------------- Sequential scan over T ----------------
  for (int t0 = 0; t0 < TSEQ; t0 += TC) {
    // Stage x chunk: x[b,i,t0..t0+TC) -> x_lds[t][b][i] (f16), coalesced bursts.
    #pragma unroll
    for (int q = 0; q < 4; ++q) {
      const int p = tid + q * 256;               // (b,i) pair id, 1024 total
      const int b = p >> 6, i = p & 63;
      const float* src = x + ((size_t)(wg * BT + b) * IDIM + i) * TSEQ + t0;
      __builtin_prefetch(src + TC, 0, 1);        // global_prefetch_b8 next chunk
      float vals[TC];
      #pragma unroll
      for (int j = 0; j < TC; j += 4) {
        const float4 v = *(const float4*)(src + j);
        vals[j] = v.x; vals[j + 1] = v.y; vals[j + 2] = v.z; vals[j + 3] = v.w;
      }
      #pragma unroll
      for (int j = 0; j < TC; ++j) x_lds[j][b][i] = (_Float16)vals[j];
    }
    __syncthreads();

    // Prologue: x-projection for step 0 of this chunk, bias folded into C.
    {
      const _Float16* xr = &x_lds[0][0][0] + rowM * XPAD;
      const v16h a0 = afrag(xr + kbA);
      const v16h a1 = afrag(xr + 32 + kbA);
      #pragma unroll
      for (int r = 0; r < 8; ++r) { xg[r] = bg_n; xp[r] = bp_n; }
      xg = wmma_f16(a0, BWgx[0], xg);
      xp = wmma_f16(a0, BWin[0], xp);
      xg = wmma_f16(a1, BWgx[1], xg);
      xp = wmma_f16(a1, BWin[1], xp);
    }

    for (int tc = 0; tc < TC; ++tc) {
      // ---- critical path: read h(buf cur), two independent 2-deep chains ----
      const _Float16* hb = &h_lds[cur][0][0] + rowM * HPAD;
      const v16h h0 = afrag(hb + kbA);
      const v16h h1 = afrag(hb + 32 + kbA);
      const v16h h2 = afrag(hb + 64 + kbA);
      const v16h h3 = afrag(hb + 96 + kbA);

      v8f g1 = wmma_f16(h0, BWgh[0], xg);       // seeded with x-proj + bias
      v8f p1 = wmma_f16(h0, BWst[0], xp);
      v8f g2 = {}, p2 = {};
      g2 = wmma_f16(h2, BWgh[2], g2);
      p2 = wmma_f16(h2, BWst[2], p2);
      g1 = wmma_f16(h1, BWgh[1], g1);
      p1 = wmma_f16(h1, BWst[1], p1);
      g2 = wmma_f16(h3, BWgh[3], g2);
      p2 = wmma_f16(h3, BWst[3], p2);

      // ---- gated update + stream stores into the other buffer ----
      _Float16* hw = &h_lds[cur ^ 1][0][0];
      #pragma unroll
      for (int r = 0; r < 8; ++r) {
        const float delta = fast_sigmoid(g1[r] + g2[r]);
        const float prop  = fast_tanh(p1[r] + p2[r]);
        hC[r] = fmaf(delta, prop - hC[r], hC[r]);
        hw[(r + hiH * 8) * HPAD + n] = (_Float16)hC[r];
      }

      // ---- look-ahead: x-projection for next step (off critical path) ----
      if (tc + 1 < TC) {
        const _Float16* xr = &x_lds[tc + 1][0][0] + rowM * XPAD;
        const v16h a0 = afrag(xr + kbA);
        const v16h a1 = afrag(xr + 32 + kbA);
        #pragma unroll
        for (int r = 0; r < 8; ++r) { xg[r] = bg_n; xp[r] = bp_n; }
        xg = wmma_f16(a0, BWgx[0], xg);
        xp = wmma_f16(a0, BWin[0], xp);
        xg = wmma_f16(a1, BWgx[1], xg);
        xp = wmma_f16(a1, BWin[1], xp);
      }

      __syncthreads();    // single barrier/step: publishes buf cur^1
      cur ^= 1;
    }
  }

  // ---------------- Output: h_final @ W_h^T + b_h  (waves 0..3) ----------------
  if (wave < 4) {
    v16h BWh[4];
    #pragma unroll
    for (int f = 0; f < 4; ++f) {
      const int kb = f * 32 + hiH * 16;
      const float* ph = W_h + (size_t)n * HD + kb;   // n in 0..63 here
      #pragma unroll
      for (int i = 0; i < 16; ++i) BWh[f][i] = (_Float16)ph[i];
    }
    const _Float16* hb = &h_lds[cur][0][0] + rowM * HPAD;
    v8f acco = {};
    #pragma unroll
    for (int f = 0; f < 4; ++f) {
      const v16h a = afrag(hb + f * 32 + kbA);
      acco = wmma_f16(a, BWh[f], acco);
    }
    const float bh_n = b_h[n];
    #pragma unroll
    for (int r = 0; r < 8; ++r) {
      const int M = r + hiH * 8;
      out[(size_t)(wg * BT + M) * ODIM + n] = acco[r] + bh_n;
    }
  }
}

extern "C" void kernel_launch(void* const* d_in, const int* in_sizes, int n_in,
                              void* d_out, int out_size, void* d_ws, size_t ws_size,
                              hipStream_t stream) {
  const float* x    = (const float*)d_in[0];
  const float* W_in = (const float*)d_in[1];
  const float* b_in = (const float*)d_in[2];
  const float* W_st = (const float*)d_in[3];
  const float* b_st = (const float*)d_in[4];
  const float* W_g  = (const float*)d_in[5];
  const float* b_g  = (const float*)d_in[6];
  const float* W_h  = (const float*)d_in[7];
  const float* b_h  = (const float*)d_in[8];
  (void)in_sizes; (void)n_in; (void)out_size; (void)d_ws; (void)ws_size;

  liquid_rnn_kernel<<<NB / BT, 256, 0, stream>>>(
      x, W_in, b_in, W_st, b_st, W_g, b_g, W_h, b_h, (float*)d_out);
}
